// LKA_LSTM_48103633715885
// MI455X (gfx1250) — matrix operand order repaired and checked by reading, not compiled, not run on hardware
//
#include <hip/hip_runtime.h>
#include <cstdint>
#include <cstddef>

// ---------------------------------------------------------------------------
// Types for CDNA5 WMMA (wave32): 16x16x32 bf16 -> f32
// ---------------------------------------------------------------------------
typedef __attribute__((ext_vector_type(16))) __bf16 v16bf;
typedef __attribute__((ext_vector_type(8)))  __bf16 v8bf;
typedef __attribute__((ext_vector_type(8)))  float  v8f;

__device__ __forceinline__ __bf16 f2bf(float f) {
  unsigned u = __builtin_bit_cast(unsigned, f);
  unsigned r = (u + 0x7fffu + ((u >> 16) & 1u)) >> 16;   // RNE
  return __builtin_bit_cast(__bf16, (unsigned short)r);
}

// CDNA5 async global->LDS copy (ASYNCcnt-tracked) + wait, via inline asm.
__device__ __forceinline__ void async_g2l_b128(unsigned lds_byte, const void* gaddr) {
  asm volatile("global_load_async_to_lds_b128 %0, %1, off"
               :: "v"(lds_byte), "v"(gaddr) : "memory");
}
__device__ __forceinline__ void wait_async0() {
  asm volatile("s_wait_asynccnt 0" ::: "memory");
}

__device__ __forceinline__ float sigm(float x) { return 1.f / (1.f + __expf(-x)); }

// ---------------------------------------------------------------------------
// Pack B (KxN, f32, arbitrary strides) into WMMA B-fragment layout:
//   dst[((nt_off+nt)*KT_total + kt_off+kt)*32 + lane][j] ,
//   lane: col = nt*16+(lane&15), kbase = kt*32+((lane&16)?8:0); element j -> k=kbase+(j<8?j:j+8)
// ---------------------------------------------------------------------------
__global__ void pack_b_kernel(const float* __restrict__ W, __bf16* __restrict__ dst,
                              int NT_loc, int KT_loc, int KT_total,
                              int nt_off, int kt_off, int k_stride, int n_stride) {
  int t = blockIdx.x * blockDim.x + threadIdx.x;
  int lane = t & 31;
  int kt = (t >> 5) % KT_loc;
  int nt = (t >> 5) / KT_loc;
  if (nt >= NT_loc) return;
  int col = nt * 16 + (lane & 15);
  int kb  = kt * 32 + ((lane & 16) ? 8 : 0);
  size_t base = (((size_t)(nt_off + nt) * KT_total) + (size_t)(kt_off + kt)) * 32 + lane;
#pragma unroll
  for (int j = 0; j < 16; ++j) {
    int kk = kb + (j < 8 ? j : j + 8);
    dst[base * 16 + j] = f2bf(W[(size_t)kk * k_stride + (size_t)col * n_stride]);
  }
}

// ---------------------------------------------------------------------------
// Pack A (MxK f32 row-major) into WMMA A-fragment layout:
//   lane: m = mt*16+(lane&15); kb8 = (lane&16)?8:0; element j -> k = kb8 + (j<8?j:j+8)
// ---------------------------------------------------------------------------
__global__ void pack_a_kernel(const float* __restrict__ A, __bf16* __restrict__ dst,
                              int MT, int KT, int lda) {
  int t = blockIdx.x * blockDim.x + threadIdx.x;
  int lane = t & 31;
  int kt = (t >> 5) % KT;
  int mt = (t >> 5) / KT;
  if (mt >= MT) return;
  int m = mt * 16 + (lane & 15);
  int kb8 = (lane & 16) ? 8 : 0;
  size_t base = ((size_t)mt * KT + kt) * 32 + lane;
#pragma unroll
  for (int j = 0; j < 16; ++j) {
    int kl = kb8 + (j < 8 ? j : j + 8);
    dst[base * 16 + j] = f2bf(A[(size_t)m * lda + (size_t)(kt * 32 + kl)]);
  }
}

// ---------------------------------------------------------------------------
// WMMA GEMM: block = 32 rows x 256 cols, 8 waves (wave owns 1 mtile x 4 ntiles).
// A tiles double-buffered in LDS via async-to-LDS; B fragments stream from L2.
// mode 0: out0 = acc + bias0            (row-major, ldc = Ncols)
// mode 1: out0 = gelu(acc + bias0)
// mode 2: QKV scatter into (B*8, 2048, 64) head-major buffers out0/out1/out2
// ---------------------------------------------------------------------------
__global__ __launch_bounds__(256)
void gemm_wmma_kernel(const __bf16* __restrict__ Ap, const __bf16* __restrict__ Bp,
                      int KT, int Ncols,
                      const float* __restrict__ bias0, const float* __restrict__ bias1,
                      const float* __restrict__ bias2,
                      float* __restrict__ out0, float* __restrict__ out1,
                      float* __restrict__ out2, int mode) {
  __shared__ __attribute__((aligned(32))) __bf16 astage[2][2][32][16];  // 4KB
  const int tid = threadIdx.x, lane = tid & 31, wid = tid >> 5;
  const int mtb = blockIdx.x * 2;
  const int ntb = blockIdx.y * 16;
  const int m_w = wid & 1;
  const int n_w = (wid >> 1) * 4;

  v8f acc[4];
#pragma unroll
  for (int t = 0; t < 4; ++t)
#pragma unroll
    for (int r = 0; r < 8; ++r) acc[t][r] = 0.f;

  // Prologue: stage k-tile 0 (threads 0..127 = waves 0..3, full EXEC)
  if (tid < 128) {
    int mi = tid >> 6, ch = tid & 63;
    unsigned l = (unsigned)(size_t)&astage[0][mi][0][0] + (unsigned)(ch * 16);
    const char* g = (const char*)(Ap + (((size_t)(mtb + mi) * KT) * 32) * 16) + ch * 16;
    async_g2l_b128(l, g);
  }
  int buf = 0;
  for (int kt = 0; kt < KT; ++kt) {
    wait_async0();
    __syncthreads();
    if (kt + 1 < KT && tid < 128) {            // prefetch next A tile into other buffer
      int mi = tid >> 6, ch = tid & 63;
      unsigned l = (unsigned)(size_t)&astage[buf ^ 1][mi][0][0] + (unsigned)(ch * 16);
      const char* g = (const char*)(Ap + (((size_t)(mtb + mi) * KT + (kt + 1)) * 32) * 16) + ch * 16;
      async_g2l_b128(l, g);
    }
    v16bf a = *(const v16bf*)&astage[buf][m_w][lane][0];
    if (kt + 1 < KT)
      __builtin_prefetch(Bp + (((size_t)(ntb + n_w) * KT + (kt + 1)) * 32 + lane) * 16, 0, 1);
#pragma unroll
    for (int t = 0; t < 4; ++t) {
      v16bf b = *(const v16bf*)(Bp + (((size_t)(ntb + n_w + t) * KT + kt) * 32 + lane) * 16);
      acc[t] = __builtin_amdgcn_wmma_f32_16x16x32_bf16(false, a, false, b, (short)0,
                                                       acc[t], false, false);
    }
    buf ^= 1;
  }

  const int mrow_lo = (mtb + m_w) * 16 + ((lane >> 4) << 3);
#pragma unroll
  for (int t = 0; t < 4; ++t) {
    int col = (ntb + n_w + t) * 16 + (lane & 15);
#pragma unroll
    for (int r = 0; r < 8; ++r) {
      int row = mrow_lo + r;
      float val = acc[t][r];
      if (mode == 2) {
        int seg = col >> 9, cc = col & 511;
        const float* bp = (seg == 0) ? bias0 : (seg == 1) ? bias1 : bias2;
        float* op = (seg == 0) ? out0 : (seg == 1) ? out1 : out2;
        val += bp[cc];
        int bi = row >> 11, n = row & 2047, hh = cc >> 6, hd = cc & 63;
        op[((size_t)(bi * 8 + hh) * 2048 + n) * 64 + hd] = val;
      } else {
        val += bias0[col];
        if (mode == 1) val = 0.5f * val * (1.f + erff(val * 0.7071067811865475f));
        out0[(size_t)row * Ncols + col] = val;
      }
    }
  }
}

// ---------------------------------------------------------------------------
// Kernelized linear attention scan: one 64-thread block per head (16 heads).
// Lane e owns v-column e; KV state (65 x col) kept in 65 VGPRs per lane.
// ---------------------------------------------------------------------------
__global__ __launch_bounds__(64)
void lka_kernel(const float* __restrict__ q, const float* __restrict__ k,
                const float* __restrict__ v, float* __restrict__ outp) {
  const int head = blockIdx.x;
  const int e = threadIdx.x;
  __shared__ float kf[65], qf[65], ksum[65];
  __shared__ float redA[64], redB[64];
  float kvr[65];
#pragma unroll
  for (int d = 0; d < 65; ++d) kvr[d] = 0.f;
  ksum[e] = 0.f;
  if (e == 0) ksum[64] = 0.f;
  __syncthreads();
  const size_t base = (size_t)head * 2048 * 64;
  for (int n = 0; n < 2048; ++n) {
    float qe = q[base + (size_t)n * 64 + e];
    float ke = k[base + (size_t)n * 64 + e];
    float ve = v[base + (size_t)n * 64 + e];
    redA[e] = qe * qe; redB[e] = ke * ke;
    __syncthreads();
    for (int s = 32; s > 0; s >>= 1) {
      if (e < s) { redA[e] += redA[e + s]; redB[e] += redB[e + s]; }
      __syncthreads();
    }
    float aq = sqrtf(redA[0]), ak = sqrtf(redB[0]);
    float sq = (1.f - exp2f(-aq)) / fmaxf(aq, 1e-12f);
    float sk = (1.f - exp2f(-ak)) / fmaxf(ak, 1e-12f);
    __syncthreads();
    qf[e] = qe * sq; kf[e] = ke * sk;
    ksum[e] += ke * sk;
    if (e == 0) { qf[64] = 1.f; kf[64] = 1.f; ksum[64] += 1.f; }
    __syncthreads();
    float part = qf[e] * ksum[e];
    if (e == 0) part += qf[64] * ksum[64];
    redA[e] = part;
    __syncthreads();
    for (int s = 32; s > 0; s >>= 1) { if (e < s) redA[e] += redA[e + s]; __syncthreads(); }
    float qk = redA[0];
    float acc = 0.f;
#pragma unroll
    for (int d = 0; d < 65; ++d) { kvr[d] += kf[d] * ve; acc += qf[d] * kvr[d]; }
    outp[base + (size_t)n * 64 + e] = acc / qk;
    __syncthreads();
  }
}

// ---------------------------------------------------------------------------
// LSTM over 2048 steps: one block, 8 waves. Per step: [x_t|h] (16x128 bf16 in
// LDS) x packed [Wih.T;Whh.T] (128x256) via 8 WMMAs/wave, gate nonlinearity,
// residual write out2 (B,N,512 layout).
// ---------------------------------------------------------------------------
__global__ __launch_bounds__(256)
void lstm_kernel(const __bf16* __restrict__ Bp, const float* __restrict__ bih,
                 const float* __restrict__ bhh, const float* __restrict__ o_in,
                 float* __restrict__ out2) {
  __shared__ __attribute__((aligned(32))) __bf16 xh[16][128];   // [x(0:64)|h(64:128)]
  __shared__ float gbuf[16][256];
  const int tid = threadIdx.x, lane = tid & 31, wid = tid >> 5;
  const int nt0 = wid * 2;

  v16bf bfr[2][4];
#pragma unroll
  for (int t = 0; t < 2; ++t)
#pragma unroll
    for (int kt = 0; kt < 4; ++kt)
      bfr[t][kt] = *(const v16bf*)(Bp + (((size_t)(nt0 + t) * 4 + kt) * 32 + lane) * 16);

  float biasv[2];
#pragma unroll
  for (int t = 0; t < 2; ++t) {
    int c = (nt0 + t) * 16 + (lane & 15);
    biasv[t] = bih[c] + bhh[c];
  }

  const int rm = tid & 15;             // sequence / matrix row
  const int cg = (tid >> 4) * 4;       // 4 owned hidden columns
  const int bb = rm >> 3, hh = rm & 7; // batch / head for output layout
  float cst[4] = {0.f, 0.f, 0.f, 0.f};
#pragma unroll
  for (int j = 0; j < 4; ++j) xh[rm][64 + cg + j] = f2bf(0.f);  // h0 = 0
  const int mA = lane & 15;
  const int kb8 = (lane & 16) ? 8 : 0;

  for (int n = 0; n < 2048; ++n) {
    float4 xin = *(const float4*)(o_in + ((size_t)rm * 2048 + n) * 64 + cg);
    xh[rm][cg + 0] = f2bf(xin.x);
    xh[rm][cg + 1] = f2bf(xin.y);
    xh[rm][cg + 2] = f2bf(xin.z);
    xh[rm][cg + 3] = f2bf(xin.w);
    __syncthreads();

    v8f acc0, acc1;
#pragma unroll
    for (int r = 0; r < 8; ++r) { acc0[r] = biasv[0]; acc1[r] = biasv[1]; }
#pragma unroll
    for (int kt = 0; kt < 4; ++kt) {
      const __bf16* ap = &xh[mA][kt * 32 + kb8];
      v8bf lo = *(const v8bf*)ap;
      v8bf hi = *(const v8bf*)(ap + 16);
      v16bf a = __builtin_shufflevector(lo, hi, 0, 1, 2, 3, 4, 5, 6, 7,
                                                8, 9, 10, 11, 12, 13, 14, 15);
      acc0 = __builtin_amdgcn_wmma_f32_16x16x32_bf16(false, a, false, bfr[0][kt], (short)0,
                                                     acc0, false, false);
      acc1 = __builtin_amdgcn_wmma_f32_16x16x32_bf16(false, a, false, bfr[1][kt], (short)0,
                                                     acc1, false, false);
    }
    const int cwr = lane & 15;
    const int mwr = (lane >> 4) << 3;
#pragma unroll
    for (int r = 0; r < 8; ++r) {
      gbuf[mwr + r][(nt0 + 0) * 16 + cwr] = acc0[r];
      gbuf[mwr + r][(nt0 + 1) * 16 + cwr] = acc1[r];
    }
    __syncthreads();

    float xv[4] = {xin.x, xin.y, xin.z, xin.w};
    size_t obase = ((size_t)bb * 2048 + n) * 512 + (size_t)hh * 64 + cg;
#pragma unroll
    for (int j = 0; j < 4; ++j) {
      int col = cg + j;
      float gi = gbuf[rm][col];
      float gf = gbuf[rm][64 + col];
      float gg = gbuf[rm][128 + col];
      float go = gbuf[rm][192 + col];
      cst[j] = sigm(gf) * cst[j] + sigm(gi) * tanhf(gg);
      float h = sigm(go) * tanhf(cst[j]);
      xh[rm][64 + col] = f2bf(h);
      out2[obase + j] = xv[j] + h;   // residual o + lstm(o)
    }
  }
}

// ---------------------------------------------------------------------------
// LayerNorm over last dim (512), one row per block.
// ---------------------------------------------------------------------------
__global__ __launch_bounds__(256)
void ln_kernel(const float* __restrict__ y, const float* __restrict__ gamma,
               const float* __restrict__ beta, float* __restrict__ out) {
  __shared__ float red[256];
  const int row = blockIdx.x, t = threadIdx.x;
  const float* yr = y + (size_t)row * 512;
  float a = yr[t], c = yr[t + 256];
  red[t] = a + c;
  __syncthreads();
  for (int s = 128; s > 0; s >>= 1) { if (t < s) red[t] += red[t + s]; __syncthreads(); }
  float mu = red[0] * (1.f / 512.f);
  __syncthreads();
  float da = a - mu, dc = c - mu;
  red[t] = da * da + dc * dc;
  __syncthreads();
  for (int s = 128; s > 0; s >>= 1) { if (t < s) red[t] += red[t + s]; __syncthreads(); }
  float inv = rsqrtf(red[0] * (1.f / 512.f) + 1e-5f);
  out[(size_t)row * 512 + t]       = da * inv * gamma[t] + beta[t];
  out[(size_t)row * 512 + t + 256] = dc * inv * gamma[t + 256] + beta[t + 256];
}

// ---------------------------------------------------------------------------
// Orchestration
// ---------------------------------------------------------------------------
extern "C" void kernel_launch(void* const* d_in, const int* in_sizes, int n_in,
                              void* d_out, int out_size, void* d_ws, size_t ws_size,
                              hipStream_t stream) {
  (void)in_sizes; (void)n_in; (void)out_size; (void)ws_size;
  const float* x    = (const float*)d_in[0];
  const float* Wq   = (const float*)d_in[1];
  const float* bq   = (const float*)d_in[2];
  const float* Wk   = (const float*)d_in[3];
  const float* bk   = (const float*)d_in[4];
  const float* Wv   = (const float*)d_in[5];
  const float* bv   = (const float*)d_in[6];
  const float* Wih  = (const float*)d_in[7];
  const float* Whh  = (const float*)d_in[8];
  const float* bih  = (const float*)d_in[9];
  const float* bhh  = (const float*)d_in[10];
  const float* W1   = (const float*)d_in[11];
  const float* b1   = (const float*)d_in[12];
  const float* W2   = (const float*)d_in[13];
  const float* b2   = (const float*)d_in[14];
  const float* gmma = (const float*)d_in[15];
  const float* beta = (const float*)d_in[16];

  char* ws = (char*)d_ws;
  const size_t MB = (size_t)1 << 20;
  __bf16* Apack = (__bf16*)(ws + 0);        // 4 MB (reused 3x)
  __bf16* Bqkv  = (__bf16*)(ws + 4 * MB);   // 1.5 MB
  __bf16* Bw1   = (__bf16*)(ws + 6 * MB);   // 0.5 MB
  __bf16* Bw2   = (__bf16*)(ws + 7 * MB);   // 0.5 MB
  __bf16* Blstm = (__bf16*)(ws + 8 * MB);   // 64 KB
  float*  qb    = (float*)(ws + 9 * MB);    // 8 MB
  float*  kbuf  = (float*)(ws + 17 * MB);   // 8 MB
  float*  vb    = (float*)(ws + 25 * MB);   // 8 MB
  float*  ob    = (float*)(ws + 33 * MB);   // 8 MB (attention out, head-major)
  float*  o2    = qb;                       // reuse after LKA
  float*  h1    = kbuf;                     // reuse
  float*  y2    = vb;                       // reuse

  // Pack weights into B-fragment bf16 layouts.
  pack_b_kernel<<<64, 256, 0, stream>>>(Wq, Bqkv, 32, 16, 16, 0,  0, 512, 1);
  pack_b_kernel<<<64, 256, 0, stream>>>(Wk, Bqkv, 32, 16, 16, 32, 0, 512, 1);
  pack_b_kernel<<<64, 256, 0, stream>>>(Wv, Bqkv, 32, 16, 16, 64, 0, 512, 1);
  pack_b_kernel<<<64, 256, 0, stream>>>(W1, Bw1,  32, 16, 16, 0,  0, 512, 1);
  pack_b_kernel<<<64, 256, 0, stream>>>(W2, Bw2,  32, 16, 16, 0,  0, 512, 1);
  // LSTM: B = [Wih.T ; Whh.T] (K=128, N=256), transposed access (k_stride=1, n_stride=64)
  pack_b_kernel<<<4, 256, 0, stream>>>(Wih, Blstm, 16, 2, 4, 0, 0, 1, 64);
  pack_b_kernel<<<4, 256, 0, stream>>>(Whh, Blstm, 16, 2, 4, 0, 2, 1, 64);

  // QKV: pack x, fused GEMM (N=1536) with per-segment bias + head-major scatter.
  pack_a_kernel<<<512, 256, 0, stream>>>(x, Apack, 256, 16, 512);
  gemm_wmma_kernel<<<dim3(128, 6), 256, 0, stream>>>(Apack, Bqkv, 16, 1536,
                                                     bq, bk, bv, qb, kbuf, vb, 2);
  // Linear-attention scan (16 heads).
  lka_kernel<<<16, 64, 0, stream>>>(qb, kbuf, vb, ob);
  // LSTM + residual, writes (B,N,512) layout.
  lstm_kernel<<<1, 256, 0, stream>>>(Blstm, bih, bhh, ob, o2);
  // MLP: gelu(o2@W1+b1)@W2+b2, then LayerNorm.
  pack_a_kernel<<<512, 256, 0, stream>>>(o2, Apack, 256, 16, 512);
  gemm_wmma_kernel<<<dim3(128, 2), 256, 0, stream>>>(Apack, Bw1, 16, 512,
                                                     b1, nullptr, nullptr,
                                                     h1, nullptr, nullptr, 1);
  pack_a_kernel<<<512, 256, 0, stream>>>(h1, Apack, 256, 16, 512);
  gemm_wmma_kernel<<<dim3(128, 2), 256, 0, stream>>>(Apack, Bw2, 16, 512,
                                                     b2, nullptr, nullptr,
                                                     y2, nullptr, nullptr, 0);
  ln_kernel<<<4096, 256, 0, stream>>>(y2, gmma, beta, (float*)d_out);
}
